// ParallelLayers_31112743092849
// MI455X (gfx1250) — matrix operand order repaired and checked by reading, not compiled, not run on hardware
//
#include <hip/hip_runtime.h>

typedef __attribute__((ext_vector_type(16))) __bf16 v16bf;
typedef __attribute__((ext_vector_type(8)))  float  v8f;

#define B_  4096
#define P_  64
#define IN_ 256
#define H1_ 512
#define H2_ 256

#define WMMA_BF16(Av, Bv, Cv) \
  __builtin_amdgcn_wmma_f32_16x16x32_bf16(false, (Av), false, (Bv), (short)0, (Cv), false, false)

// Split 16 fp32 values into hi/lo bf16 pairs (compensated bf16x3 scheme).
__device__ __forceinline__ void split_bf16(const float* f, v16bf& hi, v16bf& lo) {
#pragma unroll
  for (int i = 0; i < 16; ++i) {
    float v = f[i];
    __bf16 h = (__bf16)v;
    hi[i] = h;
    lo[i] = (__bf16)(v - (float)h);
  }
}

// Convert fp32 weights [P, K, N] into hi/lo bf16 planes stored in the exact
// per-lane order of a 32x16 bf16 B-fragment for v_wmma_f32_16x16x32_bf16:
//   tile (p, kt, nt) occupies 512 contiguous bf16 (1 KB);
//   lane t holds 16 contiguous bf16: N = nt*16 + (t&15), K = kt*32 + (t>>4)*8 + (j&7) + (j>>3)*16.
__global__ __launch_bounds__(256) void prep_w_kernel(
    const float* __restrict__ W, __bf16* __restrict__ Wh, __bf16* __restrict__ Wl,
    int K, int N) {
  int tid  = blockIdx.x * 256 + threadIdx.x;   // one thread per element
  int tile = tid >> 9;
  int r    = tid & 511;
  int lane = r >> 4;
  int j    = r & 15;
  int TN = N >> 4;
  int TK = K >> 5;
  int per_p = TK * TN;
  int p   = tile / per_p;
  int rem = tile - p * per_p;
  int kt  = rem / TN;
  int nt  = rem - kt * TN;
  int k = kt * 32 + ((lane >> 4) << 3) + (j & 7) + ((j >> 3) << 4);
  int n = (nt << 4) + (lane & 15);
  float w = W[((size_t)p * K + k) * (size_t)N + n];
  __bf16 h = (__bf16)w;
  Wh[tid] = h;
  Wl[tid] = (__bf16)(w - (float)h);
}

// Fused 2-layer per-player MLP. One block = 64 batch rows x 1 player.
// 256 threads = 8 wave32; H staged through LDS in fp32.
__global__ __launch_bounds__(256) void mlp_fused_kernel(
    const float* __restrict__ x,
    const float* __restrict__ b1,
    const float* __restrict__ b2,
    const __bf16* __restrict__ W1h, const __bf16* __restrict__ W1l,
    const __bf16* __restrict__ W2h, const __bf16* __restrict__ W2l,
    float* __restrict__ out) {
  constexpr int HPAD = H1_ + 4;          // 516 floats: +16B/row to stagger banks
  __shared__ float Hs[64 * HPAD];        // 132 KB of the 320 KB WGP LDS

  const int p     = blockIdx.y;
  const int m0    = blockIdx.x * 64;
  const int w     = threadIdx.x >> 5;
  const int lane  = threadIdx.x & 31;
  const int l16   = lane & 15;
  const int lhalf = lane >> 4;           // 0 or 1 (half-wave)
  const int mt    = w & 3;               // M-tile 0..3 (16 rows each)
  const int grp   = w >> 2;              // N half 0..1

  const v16bf* W1hv = (const v16bf*)W1h;
  const v16bf* W1lv = (const v16bf*)W1l;
  const v16bf* W2hv = (const v16bf*)W2h;
  const v16bf* W2lv = (const v16bf*)W2l;

  const v8f vzero = {0.f, 0.f, 0.f, 0.f, 0.f, 0.f, 0.f, 0.f};

  // ---------------- Layer 1: H = relu(x @ W1 + b1) ----------------
  v8f acc[16];
#pragma unroll
  for (int i = 0; i < 16; ++i) acc[i] = vzero;

  // A-fragment source row: batch row (m0 + mt*16 + l16), player p.
  const float* xrow = x + ((size_t)(m0 + mt * 16 + l16) * P_ + p) * IN_;

  for (int kt = 0; kt < IN_ / 32; ++kt) {
    const int k0 = kt * 32 + lhalf * 8;  // A layout: lane<16 -> K{0..7,16..23}
    float f[16];
    const float4 a0 = *(const float4*)(xrow + k0);
    const float4 a1 = *(const float4*)(xrow + k0 + 4);
    const float4 a2 = *(const float4*)(xrow + k0 + 16);
    const float4 a3 = *(const float4*)(xrow + k0 + 20);
    f[0]=a0.x; f[1]=a0.y; f[2]=a0.z;  f[3]=a0.w;
    f[4]=a1.x; f[5]=a1.y; f[6]=a1.z;  f[7]=a1.w;
    f[8]=a2.x; f[9]=a2.y; f[10]=a2.z; f[11]=a2.w;
    f[12]=a3.x; f[13]=a3.y; f[14]=a3.z; f[15]=a3.w;
    v16bf ah, al;
    split_bf16(f, ah, al);

    // B fragments: tiles (p, kt, nt=grp*16..+15), lane-contiguous 32 B each.
    // Software-pipelined: fragment nt+1 is in flight while WMMA(nt) issues.
    const int tbase = (((p * (IN_ / 32) + kt) * (H1_ / 16)) + grp * 16) * 32 + lane;
    v16bf bh_cur = W1hv[tbase];
    v16bf bl_cur = W1lv[tbase];
#pragma unroll
    for (int nt = 0; nt < 16; ++nt) {
      v16bf bh_nxt = bh_cur, bl_nxt = bl_cur;
      if (nt < 15) {
        bh_nxt = W1hv[tbase + (nt + 1) * 32];
        bl_nxt = W1lv[tbase + (nt + 1) * 32];
      }
      acc[nt] = WMMA_BF16(ah, bh_cur, acc[nt]);   // hi*hi
      acc[nt] = WMMA_BF16(ah, bl_cur, acc[nt]);   // hi*lo
      acc[nt] = WMMA_BF16(al, bh_cur, acc[nt]);   // lo*hi
      bh_cur = bh_nxt;
      bl_cur = bl_nxt;
    }
  }

  // bias + relu, spill H tile to LDS (fp32).
  {
    const int rbase = mt * 16 + lhalf * 8;   // C layout: VGPR v -> row v (+8 hi half)
#pragma unroll
    for (int nt = 0; nt < 16; ++nt) {
      const int ncol = (grp * 16 + nt) * 16 + l16;
      const float bias = b1[p * H1_ + ncol];
#pragma unroll
      for (int v = 0; v < 8; ++v) {
        float val = acc[nt][v] + bias;
        Hs[(rbase + v) * HPAD + ncol] = fmaxf(val, 0.f);
      }
    }
  }
  __syncthreads();

  // ---------------- Layer 2: out = relu(H @ W2 + b2) ----------------
  v8f acc2[8];
#pragma unroll
  for (int i = 0; i < 8; ++i) acc2[i] = vzero;

  const float* hrow = Hs + (mt * 16 + l16) * HPAD;

  // Prefetch A floats for kt=0.
  float f[16];
  {
    const int k0 = lhalf * 8;
    const float4 a0 = *(const float4*)(hrow + k0);
    const float4 a1 = *(const float4*)(hrow + k0 + 4);
    const float4 a2 = *(const float4*)(hrow + k0 + 16);
    const float4 a3 = *(const float4*)(hrow + k0 + 20);
    f[0]=a0.x; f[1]=a0.y; f[2]=a0.z;  f[3]=a0.w;
    f[4]=a1.x; f[5]=a1.y; f[6]=a1.z;  f[7]=a1.w;
    f[8]=a2.x; f[9]=a2.y; f[10]=a2.z; f[11]=a2.w;
    f[12]=a3.x; f[13]=a3.y; f[14]=a3.z; f[15]=a3.w;
  }

  for (int kt = 0; kt < H1_ / 32; ++kt) {
    v16bf ah, al;
    split_bf16(f, ah, al);

    // Prefetch A floats for kt+1 while this k-step computes.
    if (kt + 1 < H1_ / 32) {
      const int k0 = (kt + 1) * 32 + lhalf * 8;
      const float4 a0 = *(const float4*)(hrow + k0);
      const float4 a1 = *(const float4*)(hrow + k0 + 4);
      const float4 a2 = *(const float4*)(hrow + k0 + 16);
      const float4 a3 = *(const float4*)(hrow + k0 + 20);
      f[0]=a0.x; f[1]=a0.y; f[2]=a0.z;  f[3]=a0.w;
      f[4]=a1.x; f[5]=a1.y; f[6]=a1.z;  f[7]=a1.w;
      f[8]=a2.x; f[9]=a2.y; f[10]=a2.z; f[11]=a2.w;
      f[12]=a3.x; f[13]=a3.y; f[14]=a3.z; f[15]=a3.w;
    }

    const int tbase = (((p * (H1_ / 32) + kt) * (H2_ / 16)) + grp * 8) * 32 + lane;
    v16bf bh_cur = W2hv[tbase];
    v16bf bl_cur = W2lv[tbase];
#pragma unroll
    for (int nt = 0; nt < 8; ++nt) {
      v16bf bh_nxt = bh_cur, bl_nxt = bl_cur;
      if (nt < 7) {
        bh_nxt = W2hv[tbase + (nt + 1) * 32];
        bl_nxt = W2lv[tbase + (nt + 1) * 32];
      }
      acc2[nt] = WMMA_BF16(ah, bh_cur, acc2[nt]);
      acc2[nt] = WMMA_BF16(ah, bl_cur, acc2[nt]);
      acc2[nt] = WMMA_BF16(al, bh_cur, acc2[nt]);
      bh_cur = bh_nxt;
      bl_cur = bl_nxt;
    }
  }

  // bias + relu, store to out[b][p][n] (fp32).
  {
    const int row0 = m0 + mt * 16 + lhalf * 8;
#pragma unroll
    for (int nt = 0; nt < 8; ++nt) {
      const int ncol = (grp * 8 + nt) * 16 + l16;
      const float bias = b2[p * H2_ + ncol];
#pragma unroll
      for (int v = 0; v < 8; ++v) {
        float val = fmaxf(acc2[nt][v] + bias, 0.f);
        out[((size_t)(row0 + v) * P_ + p) * H2_ + ncol] = val;
      }
    }
  }
}

extern "C" void kernel_launch(void* const* d_in, const int* in_sizes, int n_in,
                              void* d_out, int out_size, void* d_ws, size_t ws_size,
                              hipStream_t stream) {
  (void)in_sizes; (void)n_in; (void)out_size; (void)ws_size;
  const float* x  = (const float*)d_in[0];
  const float* W1 = (const float*)d_in[1];
  const float* b1 = (const float*)d_in[2];
  const float* W2 = (const float*)d_in[3];
  const float* b2 = (const float*)d_in[4];
  float* out = (float*)d_out;

  const size_t w1e = (size_t)P_ * IN_ * H1_;   // 8,388,608 elements
  const size_t w2e = (size_t)P_ * H1_ * H2_;   // 8,388,608 elements
  __bf16* W1h = (__bf16*)d_ws;                 // 4 planes * 16 MB = 64 MB scratch
  __bf16* W1l = W1h + w1e;
  __bf16* W2h = W1l + w1e;
  __bf16* W2l = W2h + w2e;

  prep_w_kernel<<<(int)(w1e / 256), 256, 0, stream>>>(W1, W1h, W1l, IN_, H1_);
  prep_w_kernel<<<(int)(w2e / 256), 256, 0, stream>>>(W2, W2h, W2l, H1_, H2_);

  dim3 grid(B_ / 64, P_);
  mlp_fused_kernel<<<grid, 256, 0, stream>>>(x, b1, b2, W1h, W1l, W2h, W2l, out);
}
